// SSM_black_con_56994216018399
// MI455X (gfx1250) — matrix operand order repaired and checked by reading, not compiled, not run on hardware
//
#include <hip/hip_runtime.h>

// ---------------- problem constants ----------------
#define NXd   64
#define NYd   32
#define NMd   32
#define NDTd  32
#define NUd   16
#define NDd   16
#define NHd   256
#define NESTd 64
#define NHORd 256
#define BATCHd 2048
#define DXMIN_ (-0.5f)
#define DXMAX_ (0.5f)

typedef __bf16 bf16_t;
typedef __attribute__((ext_vector_type(16))) __bf16 v16bf;
typedef __attribute__((ext_vector_type(8)))  float  v8f;

// ---------------- workspace layout ----------------
// f32 region (element offsets)
enum : int {
  F_AW = 0,        // 64x64
  F_CW = 4096,     // 64x32
  F_EW = 6144,     // 16x64
  F_P   = 8192,    // 64x64
  F_T1  = 12288,   // 64x64
  F_PP  = 16384,   // 64x64
  F_CTP = 20480,   // 32x64
  F_AUG = 22528,   // 32x64
  F_PC  = 24576,   // 64x32
  F_L   = 26624,   // 64x32
  F_TOTAL = 28672
};
#define WSF_BYTES (128*1024)
// bf16 region (element offsets), starts at d_ws + WSF_BYTES.
// All matrices stored PRE-SWIZZLED into 512-element B-fragment tiles (32Kx16N):
// element (tile, lane, e): k_in = (e>=8)*16 + (lane>=16)*8 + (e&7), n_in = lane&15.
enum : int {
  OFF_BA   = 0,       // A_W   64x64   tiles [nt][kt], KT=2, NT=4  -> 8 tiles
  OFF_BC   = 4096,    // C_W   64x32   KT=2, NT=2 -> 4 tiles
  OFF_BEP  = 6144,    // E_W   32x64 (k>=16 zero) KT=1, NT=4 -> 4 tiles
  OFF_BBP  = 8192,    // B_W   32x64 (k>=16 zero) KT=1, NT=4 -> 4 tiles
  OFF_BW1T = 10240,   // W1^T  64x256  KT=2, NT=16 -> 32 tiles
  OFF_BW2T = 26624,   // W2^T  256x16  KT=8, NT=1 -> 8 tiles
  OFF_BLT  = 30720    // 64 steps of L^T (32x64), KT=1, NT=4 -> 4 tiles/step
};

// decode swizzled element index -> (tile coords, in-tile k, in-tile n)
static __device__ __forceinline__ void sw_decode(int idx, int KT, int& kt, int& nt, int& ki, int& ni) {
  int tile = idx >> 9;
  kt = tile % KT; nt = tile / KT;
  int e9 = idx & 511;
  int lane = e9 >> 4, e = e9 & 15;
  ni = lane & 15;
  ki = ((e >> 3) << 4) + ((lane >> 4) << 3) + (e & 7);
}

// ---------------- WMMA helpers (CDNA5 layouts, wave32) ----------------
static __device__ __forceinline__ v8f wmma_bf(v16bf a, v16bf b, v8f c) {
  return __builtin_amdgcn_wmma_f32_16x16x32_bf16(false, a, false, b, (short)0, c, false, false);
}

// A fragment 16x32 bf16 from f32 source (global or LDS)
static __device__ __forceinline__ v16bf ldA_f32(const float* src, int ld, int k0) {
  const int lane = threadIdx.x & 31, m = lane & 15, h = (lane >> 4) & 1;
  v16bf a;
#pragma unroll
  for (int v = 0; v < 8; ++v) {
    int kb = k0 + ((v & 4) ? 16 : 0) + ((v & 3) << 1) + (h << 3);
    a[2*v]   = (bf16_t)src[m*ld + kb];
    a[2*v+1] = (bf16_t)src[m*ld + kb + 1];
  }
  return a;
}
static __device__ __forceinline__ v16bf ldA_f32_pad(const float* src, int ld, int width) {
  const int lane = threadIdx.x & 31, m = lane & 15, h = (lane >> 4) & 1;
  v16bf a;
#pragma unroll
  for (int v = 0; v < 8; ++v) {
    int kb = ((v & 4) ? 16 : 0) + ((v & 3) << 1) + (h << 3);
    a[2*v]   = (kb     < width) ? (bf16_t)src[m*ld + kb]     : (bf16_t)0.f;
    a[2*v+1] = (kb + 1 < width) ? (bf16_t)src[m*ld + kb + 1] : (bf16_t)0.f;
  }
  return a;
}
static __device__ __forceinline__ v16bf ldA_bf16(const bf16_t* src, int ld, int k0) {
  const int lane = threadIdx.x & 31, m = lane & 15, h = (lane >> 4) & 1;
  v16bf a;
#pragma unroll
  for (int v = 0; v < 8; ++v) {
    int kb = k0 + ((v & 4) ? 16 : 0) + ((v & 3) << 1) + (h << 3);
    a[2*v]   = src[m*ld + kb];
    a[2*v+1] = src[m*ld + kb + 1];
  }
  return a;
}
// residual (ym - y), both 16x32 f32 in LDS
static __device__ __forceinline__ v16bf ldA_resid(const float* ymL, const float* ylds) {
  const int lane = threadIdx.x & 31, m = lane & 15, h = (lane >> 4) & 1;
  v16bf a;
#pragma unroll
  for (int v = 0; v < 8; ++v) {
    int kb = ((v & 4) ? 16 : 0) + ((v & 3) << 1) + (h << 3);
    a[2*v]   = (bf16_t)(ymL[m*32 + kb]     - ylds[m*32 + kb]);
    a[2*v+1] = (bf16_t)(ymL[m*32 + kb + 1] - ylds[m*32 + kb + 1]);
  }
  return a;
}
// B fragment from pre-swizzled storage: lane reads 16 contiguous bf16 (32B)
static __device__ __forceinline__ v16bf ldB_sw(const bf16_t* base, int tile) {
  const int lane = threadIdx.x & 31;
  return *(const v16bf*)(base + tile*512 + lane*16);
}
// C fragment f32 16x16: VGPR r -> (M = r + (lane>=16?8:0), N = lane&15)
static __device__ __forceinline__ void stC_lds_f32(float* dst, int ld, int n0, v8f c) {
  const int lane = threadIdx.x & 31, n = lane & 15, mo = (lane & 16) ? 8 : 0;
#pragma unroll
  for (int r = 0; r < 8; ++r) dst[(r + mo)*ld + n0 + n] = c[r];
}
static __device__ __forceinline__ void addC_lds_f32(float* dst, int ld, int n0, v8f c) {
  const int lane = threadIdx.x & 31, n = lane & 15, mo = (lane & 16) ? 8 : 0;
#pragma unroll
  for (int r = 0; r < 8; ++r) dst[(r + mo)*ld + n0 + n] += c[r];
}
static __device__ __forceinline__ void stC_lds_bf16_relu(bf16_t* dst, int ld, int n0, v8f c) {
  const int lane = threadIdx.x & 31, n = lane & 15, mo = (lane & 16) ? 8 : 0;
#pragma unroll
  for (int r = 0; r < 8; ++r) dst[(r + mo)*ld + n0 + n] = (bf16_t)fmaxf(c[r], 0.f);
}

// ---------------- async global->LDS staging (gfx1250 TDM-lite path) ----------------
// Each lane DMAs 16B per issue; BYTES must be a multiple of 512 (32 lanes * 16B).
template <int BYTES>
static __device__ __forceinline__ void stage_async(const float* __restrict__ g, float* lds) {
  const int lane = threadIdx.x & 31;
  unsigned loff = (unsigned)(uintptr_t)lds + (unsigned)(lane * 16);
  unsigned goff = (unsigned)(lane * 16);
#pragma unroll
  for (int c = 0; c < BYTES; c += 512) {
    asm volatile("global_load_async_to_lds_b128 %0, %1, %2"
                 :: "v"(loff + c), "v"(goff + c), "s"(g)
                 : "memory");
  }
}
template <int N>
static __device__ __forceinline__ void wait_async() {
  asm volatile("s_wait_asynccnt %0" :: "i"(N) : "memory");
}

// per-wave input staging buffer (double buffered)
struct __align__(16) Stage {
  float mf[16*NMd];    // 2KB : 4 async issues
  float dt[16*NDTd];   // 2KB : 4
  float ym[16*NYd];    // 2KB : 4 (estimation only)
  float dp[16*NDd];    // 1KB : 2
};
#define EST_ISSUES 14
#define HOR_ISSUES 10

// ---------------- prep kernel ----------------
static __device__ __forceinline__ float sigm_(float x) { return 1.f / (1.f + expf(-x)); }

static __device__ float orth_err_(const float* __restrict__ M, int n, float* red, int tid) {
  float s1 = 0.f, s2 = 0.f;
  for (int idx = tid; idx < n*n; idx += 1024) {
    int i = idx / n, j = idx - i*n;
    float d1 = 0.f, d2 = 0.f;
    for (int k = 0; k < n; ++k) {
      d1 = fmaf(M[i*n + k], M[j*n + k], d1);
      d2 = fmaf(M[k*n + i], M[k*n + j], d2);
    }
    float e = (i == j) ? 1.f : 0.f;
    float t1 = e - d1, t2 = e - d2;
    s1 = fmaf(t1, t1, s1);
    s2 = fmaf(t2, t2, s2);
  }
  __syncthreads();
  red[tid] = s1; __syncthreads();
  for (int o = 512; o > 0; o >>= 1) { if (tid < o) red[tid] += red[tid + o]; __syncthreads(); }
  float r1 = red[0]; __syncthreads();
  red[tid] = s2; __syncthreads();
  for (int o = 512; o > 0; o >>= 1) { if (tid < o) red[tid] += red[tid + o]; __syncthreads(); }
  float r2 = red[0]; __syncthreads();
  return sqrtf(r1) + sqrtf(r2);
}

__global__ __launch_bounds__(1024) void ssm_prep(
    const float* __restrict__ U_A, const float* __restrict__ V_A, const float* __restrict__ sigma_A,
    const float* __restrict__ U_C, const float* __restrict__ V_C, const float* __restrict__ sigma_C,
    const float* __restrict__ E_w, const float* __restrict__ E_s, const float* __restrict__ B_w,
    const float* __restrict__ W1, const float* __restrict__ W2,
    float* __restrict__ wsf, bf16_t* __restrict__ wsb, float* __restrict__ outSpect)
{
  const int tid = threadIdx.x;
  const int NT = 1024;
  __shared__ float red[1024];
  __shared__ float sA[64], sC[64];
  __shared__ float colk[32];
  __shared__ float pivinv;

  float* AW = wsf + F_AW;  float* CW = wsf + F_CW;  float* EW = wsf + F_EW;
  float* P  = wsf + F_P;   float* T1 = wsf + F_T1;  float* Pp = wsf + F_PP;
  float* CtP = wsf + F_CTP; float* AUG = wsf + F_AUG;
  float* PC = wsf + F_PC;  float* L  = wsf + F_L;
  bf16_t* bA   = wsb + OFF_BA;   bf16_t* bC   = wsb + OFF_BC;
  bf16_t* bEp  = wsb + OFF_BEP;  bf16_t* bBp  = wsb + OFF_BBP;
  bf16_t* bW1T = wsb + OFF_BW1T; bf16_t* bW2T = wsb + OFF_BW2T;
  bf16_t* bLT  = wsb + OFF_BLT;

  if (tid < 64) {
    sA[tid] = 1.0f - 0.4f * sigm_(sigma_A[tid]);   // smin=0.6, smax=1.0
    sC[tid] = 1.0f - 0.1f * sigm_(sigma_C[tid]);   // smin=0.9, smax=1.0
  }
  __syncthreads();

  // A_W = U_A @ diag(sA) @ V_A ; C_W = U_C[:,:32] @ diag(sC[:32]) @ V_C ; E_W scaled softmax
  for (int idx = tid; idx < 4096; idx += NT) {
    int i = idx >> 6, j = idx & 63;
    float acc = 0.f;
    for (int k = 0; k < 64; ++k) acc = fmaf(U_A[i*64 + k] * sA[k], V_A[k*64 + j], acc);
    AW[idx] = acc;
  }
  for (int idx = tid; idx < 2048; idx += NT) {
    int i = idx >> 5, j = idx & 31;
    float acc = 0.f;
    for (int k = 0; k < 32; ++k) acc = fmaf(U_C[i*64 + k] * sC[k], V_C[k*32 + j], acc);
    CW[idx] = acc;
  }
  if (tid < 16) {
    int i = tid;
    float mx = -1e30f;
    for (int j = 0; j < 64; ++j) mx = fmaxf(mx, E_w[i*64 + j]);
    float sum = 0.f;
    for (int j = 0; j < 64; ++j) sum += expf(E_w[i*64 + j] - mx);
    for (int j = 0; j < 64; ++j) {
      float s = 1.0f - 0.95f * sigm_(E_s[i*64 + j]);   // smin=0.05, smax=1.0
      EW[i*64 + j] = s * expf(E_w[i*64 + j] - mx) / sum;
    }
  }
  __syncthreads();

  // ---- swizzled bf16 weight tiles ----
  for (int idx = tid; idx < 4096; idx += NT) {          // bA  (KT=2)
    int kt, nt, ki, ni; sw_decode(idx, 2, kt, nt, ki, ni);
    bA[idx] = (bf16_t)AW[(kt*32 + ki)*64 + nt*16 + ni];
  }
  for (int idx = tid; idx < 2048; idx += NT) {          // bC  (KT=2)
    int kt, nt, ki, ni; sw_decode(idx, 2, kt, nt, ki, ni);
    bC[idx] = (bf16_t)CW[(kt*32 + ki)*32 + nt*16 + ni];
  }
  for (int idx = tid; idx < 2048; idx += NT) {          // bEp (KT=1, pad k>=16)
    int kt, nt, ki, ni; sw_decode(idx, 1, kt, nt, ki, ni);
    bEp[idx] = (ki < 16) ? (bf16_t)EW[ki*64 + nt*16 + ni] : (bf16_t)0.f;
  }
  for (int idx = tid; idx < 2048; idx += NT) {          // bBp (KT=1, pad k>=16)
    int kt, nt, ki, ni; sw_decode(idx, 1, kt, nt, ki, ni);
    bBp[idx] = (ki < 16) ? (bf16_t)fmaxf(B_w[ki*64 + nt*16 + ni], 0.f) : (bf16_t)0.f;
  }
  for (int idx = tid; idx < 16384; idx += NT) {         // bW1T (KT=2): W1^T[k][n] = W1[n][k]
    int kt, nt, ki, ni; sw_decode(idx, 2, kt, nt, ki, ni);
    bW1T[idx] = (bf16_t)W1[(nt*16 + ni)*64 + kt*32 + ki];
  }
  for (int idx = tid; idx < 4096; idx += NT) {          // bW2T (KT=8): W2^T[k][n] = W2[n][k]
    int kt, nt, ki, ni; sw_decode(idx, 8, kt, nt, ki, ni);
    bW2T[idx] = (bf16_t)W2[ni*256 + kt*32 + ki];
  }
  __syncthreads();

  // spectral/orthogonality error
  float serr = orth_err_(U_A, 64, red, tid) + orth_err_(V_A, 64, red, tid)
             + orth_err_(U_C, 64, red, tid) + orth_err_(V_C, 32, red, tid);
  for (int idx = tid; idx < NHORd; idx += NT) outSpect[idx] = serr;

  // ---- precompute Kalman gains L_t (batch-independent) ----
  for (int idx = tid; idx < 4096; idx += NT) P[idx] = ((idx >> 6) == (idx & 63)) ? 1.f : 0.f;
  for (int idx = tid; idx < 2048; idx += NT) bLT[idx] = (bf16_t)0.f;   // L used at step 0 is zeros
  __syncthreads();

  for (int t = 0; t < 63; ++t) {
    for (int idx = tid; idx < 4096; idx += NT) {   // T1 = A @ P
      int i = idx >> 6, j = idx & 63; float a = 0.f;
      for (int k = 0; k < 64; ++k) a = fmaf(AW[i*64 + k], P[k*64 + j], a);
      T1[idx] = a;
    }
    __syncthreads();
    for (int idx = tid; idx < 4096; idx += NT) {   // Pp = T1 @ A^T + I
      int i = idx >> 6, j = idx & 63; float a = (i == j) ? 1.f : 0.f;
      for (int k = 0; k < 64; ++k) a = fmaf(T1[i*64 + k], AW[j*64 + k], a);
      Pp[idx] = a;
    }
    __syncthreads();
    for (int idx = tid; idx < 2048; idx += NT) {   // CtP = C^T @ Pp (32x64)
      int i = idx >> 6, j = idx & 63; float a = 0.f;
      for (int k = 0; k < 64; ++k) a = fmaf(CW[k*32 + i], Pp[k*64 + j], a);
      CtP[idx] = a;
    }
    __syncthreads();
    for (int idx = tid; idx < 2048; idx += NT) {   // AUG = [ I + CtP@C | I ]
      int i = idx >> 6, j = idx & 63; float a;
      if (j < 32) { a = (i == j) ? 1.f : 0.f; for (int k = 0; k < 64; ++k) a = fmaf(CtP[i*64 + k], CW[k*32 + j], a); }
      else a = (i == (j - 32)) ? 1.f : 0.f;
      AUG[idx] = a;
    }
    __syncthreads();
    for (int k = 0; k < 32; ++k) {                 // Gauss-Jordan
      if (tid < 32) colk[tid] = AUG[tid*64 + k];
      if (tid == 0) pivinv = 1.f / AUG[k*64 + k];
      __syncthreads();
      if (tid < 64) AUG[k*64 + tid] *= pivinv;
      __syncthreads();
      for (int idx = tid; idx < 2048; idx += NT) {
        int i = idx >> 6, j = idx & 63;
        if (i != k) AUG[idx] -= colk[i] * AUG[k*64 + j];
      }
      __syncthreads();
    }
    for (int idx = tid; idx < 2048; idx += NT) {   // PC = Pp @ C (64x32)
      int i = idx >> 5, j = idx & 31; float a = 0.f;
      for (int k = 0; k < 64; ++k) a = fmaf(Pp[i*64 + k], CW[k*32 + j], a);
      PC[idx] = a;
    }
    __syncthreads();
    for (int idx = tid; idx < 2048; idx += NT) {   // L = PC @ Sinv (64x32)
      int i = idx >> 5, j = idx & 31; float a = 0.f;
      for (int k = 0; k < 32; ++k) a = fmaf(PC[i*32 + k], AUG[k*64 + 32 + j], a);
      L[idx] = a;
    }
    __syncthreads();
    for (int idx = tid; idx < 2048; idx += NT) {   // bLT[t+1] swizzled: LT[k][n] = L[n][k]
      int kt, nt, ki, ni; sw_decode(idx, 1, kt, nt, ki, ni);
      bLT[(t + 1)*2048 + idx] = (bf16_t)L[(nt*16 + ni)*32 + ki];
    }
    for (int idx = tid; idx < 4096; idx += NT) {   // P = I - L @ CtP
      int i = idx >> 6, j = idx & 63; float a = (i == j) ? 1.f : 0.f;
      for (int k = 0; k < 32; ++k) a = fmaf(-L[i*32 + k], CtP[k*64 + j], a);
      P[idx] = a;
    }
    __syncthreads();
  }
}

// ---------------- main fused sequence kernel ----------------
// 64 blocks x 64 threads; each wave32 owns a 16-row batch tile, runs full sequence.
__global__ __launch_bounds__(64) void ssm_main(
    const float* __restrict__ Ym, const float* __restrict__ Mfp, const float* __restrict__ DTp,
    const float* __restrict__ Dp, const float* __restrict__ Mf, const float* __restrict__ DTh,
    const float* __restrict__ Dh, const float* __restrict__ Xmn, const float* __restrict__ Xmx,
    const float* __restrict__ Umn, const float* __restrict__ Umx,
    const float* __restrict__ x0,
    const bf16_t* __restrict__ wsb, float* __restrict__ out)
{
  __shared__ float  xf_s[2][16*64];
  __shared__ float  yf_s[2][16*32];
  __shared__ bf16_t hb_s[2][16*256];
  __shared__ bf16_t ub_s[2][16*32];
  __shared__ Stage  stg_s[2][2];   // [buf][wave]
  const int wave = threadIdx.x >> 5;
  const int lane = threadIdx.x & 31;
  float*  xf = xf_s[wave];
  float*  yf = yf_s[wave];
  bf16_t* hb = hb_s[wave];
  bf16_t* ub = ub_s[wave];
  const int row0 = (blockIdx.x * 2 + wave) * 16;

  const bf16_t* bA   = wsb + OFF_BA;
  const bf16_t* bC   = wsb + OFF_BC;
  const bf16_t* bEp  = wsb + OFF_BEP;
  const bf16_t* bBp  = wsb + OFF_BBP;
  const bf16_t* bW1T = wsb + OFF_BW1T;
  const bf16_t* bW2T = wsb + OFF_BW2T;
  const bf16_t* bLT  = wsb + OFF_BLT;

  // x0 broadcast; zero the K-padding half of the u buffer once (only cols<16 ever rewritten)
  for (int i = lane; i < 16*64; i += 32) xf[i] = x0[i & 63];
  for (int i = lane; i < 16*32; i += 32) ub[i] = (bf16_t)0.f;

  const int nn = lane & 15;
  const int mo = (lane & 16) ? 8 : 0;

  // async staging issue helpers
  auto issue_est = [&](int t, int buf) {
    Stage& s = stg_s[buf][wave];
    stage_async<2048>(Mfp + ((size_t)t*BATCHd + row0) * NMd,  s.mf);
    stage_async<2048>(DTp + ((size_t)t*BATCHd + row0) * NDTd, s.dt);
    stage_async<2048>(Ym  + ((size_t)t*BATCHd + row0) * NYd,  s.ym);
    stage_async<1024>(Dp  + ((size_t)t*BATCHd + row0) * NDd,  s.dp);
  };
  auto issue_hor = [&](int t, int buf) {
    Stage& s = stg_s[buf][wave];
    stage_async<2048>(Mf  + ((size_t)t*BATCHd + row0) * NMd,  s.mf);
    stage_async<2048>(DTh + ((size_t)t*BATCHd + row0) * NDTd, s.dt);
    stage_async<1024>(Dh  + ((size_t)t*BATCHd + row0) * NDd,  s.dp);
  };

  issue_est(0, 0);

  // ================= estimation (64 steps) =================
  for (int t = 0; t < NESTd; ++t) {
    if (t + 1 < NESTd) { issue_est(t + 1, (t + 1) & 1); wait_async<EST_ISSUES>(); }
    else               { issue_hor(0, 0);               wait_async<HOR_ISSUES>(); }
    Stage& st = stg_s[t & 1][wave];

    // policy hidden: H = relu([mf,dt] @ W1^T)
    v16bf aZ0 = ldA_f32(st.mf, NMd, 0);
    v16bf aZ1 = ldA_f32(st.dt, NDTd, 0);
#pragma unroll
    for (int j = 0; j < 16; ++j) {
      v8f c = {};
      c = wmma_bf(aZ0, ldB_sw(bW1T, j*2 + 0), c);
      c = wmma_bf(aZ1, ldB_sw(bW1T, j*2 + 1), c);
      stC_lds_bf16_relu(hb, NHd, j*16, c);
    }
    // u = relu(H @ W2^T)
    {
      v8f c = {};
#pragma unroll
      for (int kt = 0; kt < 8; ++kt)
        c = wmma_bf(ldA_bf16(hb, NHd, kt*32), ldB_sw(bW2T, kt), c);
      stC_lds_bf16_relu(ub, 32, 0, c);
    }
    // x = x@A + u@B + d@E
    v16bf aU  = ldA_bf16(ub, 32, 0);
    v16bf aD  = ldA_f32_pad(st.dp, NDd, NDd);
    v16bf aX0 = ldA_f32(xf, 64, 0);
    v16bf aX1 = ldA_f32(xf, 64, 32);
#pragma unroll
    for (int j = 0; j < 4; ++j) {
      v8f c = {};
      c = wmma_bf(aX0, ldB_sw(bA, j*2 + 0), c);
      c = wmma_bf(aX1, ldB_sw(bA, j*2 + 1), c);
      c = wmma_bf(aU,  ldB_sw(bBp, j), c);
      c = wmma_bf(aD,  ldB_sw(bEp, j), c);
      stC_lds_f32(xf, 64, j*16, c);
    }
    // y = x @ C
    v16bf aN0 = ldA_f32(xf, 64, 0);
    v16bf aN1 = ldA_f32(xf, 64, 32);
#pragma unroll
    for (int j = 0; j < 2; ++j) {
      v8f c = {};
      c = wmma_bf(aN0, ldB_sw(bC, j*2 + 0), c);
      c = wmma_bf(aN1, ldB_sw(bC, j*2 + 1), c);
      stC_lds_f32(yf, NYd, j*16, c);
    }
    // x += (ym - y) @ L_t^T
    v16bf aR = ldA_resid(st.ym, yf);
#pragma unroll
    for (int j = 0; j < 4; ++j) {
      v8f c = {};
      c = wmma_bf(aR, ldB_sw(bLT, t*4 + j), c);
      addC_lds_f32(xf, 64, j*16, c);
    }
  }

  // ================= horizon rollout (256 steps) =================
  const size_t S = (size_t)NHORd * BATCHd;
  float* oX   = out;
  float* oY   = oX   + S*64;
  float* oU   = oY   + S*32;
  float* oSxm = oU   + S*16;
  float* oSxM = oSxm + S*64;
  float* oSum = oSxM + S*64;
  float* oSuM = oSum + S*16;
  float* oSdx = oSuM + S*16;
  float* oSdu = oSdx + S*64;
  float* oSdd = oSdu + S*64;

  for (int t = 0; t < NHORd; ++t) {
    if (t + 1 < NHORd) { issue_hor(t + 1, (t + 1) & 1); wait_async<HOR_ISSUES>(); }
    else               { wait_async<0>(); }
    Stage& st = stg_s[t & 1][wave];

    const size_t rb = (size_t)t*BATCHd + row0;
    const float* xmn = Xmn + rb * NXd;
    const float* xmx = Xmx + rb * NXd;
    const float* umn = Umn + rb * NUd;
    const float* umx = Umx + rb * NUd;
    if (t + 1 < NHORd) {
      __builtin_prefetch(xmn + (size_t)BATCHd*NXd, 0, 1);
      __builtin_prefetch(xmx + (size_t)BATCHd*NXd, 0, 1);
      __builtin_prefetch(umn + (size_t)BATCHd*NUd, 0, 1);
      __builtin_prefetch(umx + (size_t)BATCHd*NUd, 0, 1);
    }

    // policy
    v16bf aZ0 = ldA_f32(st.mf, NMd, 0);
    v16bf aZ1 = ldA_f32(st.dt, NDTd, 0);
#pragma unroll
    for (int j = 0; j < 16; ++j) {
      v8f c = {};
      c = wmma_bf(aZ0, ldB_sw(bW1T, j*2 + 0), c);
      c = wmma_bf(aZ1, ldB_sw(bW1T, j*2 + 1), c);
      stC_lds_bf16_relu(hb, NHd, j*16, c);
    }
    v8f cu = {};
#pragma unroll
    for (int kt = 0; kt < 8; ++kt)
      cu = wmma_bf(ldA_bf16(hb, NHd, kt*32), ldB_sw(bW2T, kt), cu);
    // u outputs + slacks, directly from accumulator fragment
#pragma unroll
    for (int r = 0; r < 8; ++r) {
      int m = r + mo;
      float u = fmaxf(cu[r], 0.f);
      ub[m*32 + nn] = (bf16_t)u;
      size_t o = (rb + m)*NUd + nn;
      oU[o]   = u;
      oSum[o] = fmaxf(0.f, umn[m*NUd + nn] - u);
      oSuM[o] = fmaxf(0.f, u - umx[m*NUd + nn]);
    }
    // Bu, Ed (+ their slack outputs)
    v16bf aU = ldA_bf16(ub, 32, 0);
    v16bf aD = ldA_f32_pad(st.dp, NDd, NDd);
    v8f cBu[4], cEd[4];
#pragma unroll
    for (int j = 0; j < 4; ++j) {
      v8f c = {}; c = wmma_bf(aU, ldB_sw(bBp, j), c); cBu[j] = c;
      v8f e = {}; e = wmma_bf(aD, ldB_sw(bEp, j), e); cEd[j] = e;
#pragma unroll
      for (int r = 0; r < 8; ++r) {
        int m = r + mo;
        size_t o = (rb + m)*NXd + j*16 + nn;
        float b = c[r], ed = e[r];
        oSdu[o] = fmaxf(0.f, -b + DXMIN_) + fmaxf(0.f, b - DXMAX_);
        oSdd[o] = fmaxf(0.f, -ed + DXMIN_) + fmaxf(0.f, ed - DXMAX_);
      }
    }
    // x_new = x@A + Bu + Ed; X/Sxmin/Sxmax/Sdx_x outputs; update state
    v16bf aX0 = ldA_f32(xf, 64, 0);
    v16bf aX1 = ldA_f32(xf, 64, 32);
#pragma unroll
    for (int j = 0; j < 4; ++j) {
      v8f c = cBu[j] + cEd[j];
      c = wmma_bf(aX0, ldB_sw(bA, j*2 + 0), c);
      c = wmma_bf(aX1, ldB_sw(bA, j*2 + 1), c);
#pragma unroll
      for (int r = 0; r < 8; ++r) {
        int m = r + mo, col = j*16 + nn;
        float xn = c[r];
        float xo = xf[m*64 + col];
        size_t o = (rb + m)*NXd + col;
        oX[o]   = xn;
        oSdx[o] = xn - xo;
        oSxm[o] = fmaxf(0.f, xmn[m*NXd + col] - xn);
        oSxM[o] = fmaxf(0.f, xn - xmx[m*NXd + col]);
        xf[m*64 + col] = xn;
      }
    }
    // y = x_new @ C
    v16bf aN0 = ldA_f32(xf, 64, 0);
    v16bf aN1 = ldA_f32(xf, 64, 32);
#pragma unroll
    for (int j = 0; j < 2; ++j) {
      v8f c = {};
      c = wmma_bf(aN0, ldB_sw(bC, j*2 + 0), c);
      c = wmma_bf(aN1, ldB_sw(bC, j*2 + 1), c);
#pragma unroll
      for (int r = 0; r < 8; ++r) { int m = r + mo; oY[(rb + m)*NYd + j*16 + nn] = c[r]; }
    }
  }
}

// ---------------- host entry ----------------
extern "C" void kernel_launch(void* const* d_in, const int* in_sizes, int n_in,
                              void* d_out, int out_size, void* d_ws, size_t ws_size,
                              hipStream_t stream) {
  (void)in_sizes; (void)n_in; (void)out_size; (void)ws_size;
  const float* Ym     = (const float*)d_in[0];
  const float* Mfp    = (const float*)d_in[1];
  const float* DTp    = (const float*)d_in[2];
  const float* Dp     = (const float*)d_in[3];
  const float* Mf     = (const float*)d_in[4];
  const float* DTh    = (const float*)d_in[5];
  const float* Dh     = (const float*)d_in[6];
  const float* XMIN   = (const float*)d_in[7];
  const float* XMAX   = (const float*)d_in[8];
  const float* UMIN   = (const float*)d_in[9];
  const float* UMAX   = (const float*)d_in[10];
  const float* U_A    = (const float*)d_in[11];
  const float* V_A    = (const float*)d_in[12];
  const float* sigA   = (const float*)d_in[13];
  const float* U_C    = (const float*)d_in[14];
  const float* V_C    = (const float*)d_in[15];
  const float* sigC   = (const float*)d_in[16];
  const float* E_w    = (const float*)d_in[17];
  const float* E_s    = (const float*)d_in[18];
  const float* B_w    = (const float*)d_in[19];
  const float* W1     = (const float*)d_in[20];
  const float* W2     = (const float*)d_in[21];
  const float* x0     = (const float*)d_in[22];

  float*  wsf = (float*)d_ws;
  bf16_t* wsb = (bf16_t*)((char*)d_ws + WSF_BYTES);
  float*  out = (float*)d_out;
  const size_t S = (size_t)NHORd * BATCHd;
  float* outSpect = out + S * (64 + 32 + 16 + 64 + 64 + 16 + 16 + 64 + 64 + 64);

  ssm_prep<<<1, 1024, 0, stream>>>(U_A, V_A, sigA, U_C, V_C, sigC,
                                   E_w, E_s, B_w, W1, W2, wsf, wsb, outSpect);
  ssm_main<<<64, 64, 0, stream>>>(Ym, Mfp, DTp, Dp, Mf, DTh, Dh,
                                  XMIN, XMAX, UMIN, UMAX, x0, wsb, out);
}
